// LSTM_91199335563712
// MI455X (gfx1250) — matrix operand order repaired and checked by reading, not compiled, not run on hardware
//
#include <hip/hip_runtime.h>
#include <hip/hip_bf16.h>
#include <math.h>

// ---------------- problem constants ----------------
constexpr int kH   = 256;   // hidden
constexpr int kB   = 256;   // batch
constexpr int kT   = 784;   // seq len
constexpr int kC   = 10;    // classes
constexpr int kNWG = 16;    // persistent workgroups (one 16-row M-tile each)
constexpr int kTPB = 256;   // 8 waves (wave32)

// ---------------- WMMA vector types ----------------
typedef __attribute__((ext_vector_type(16))) __bf16 v16bf;
typedef __attribute__((ext_vector_type(8)))  __bf16 v8bf;
typedef __attribute__((ext_vector_type(8)))  float  v8f;

union V16U { v16bf v; v8bf h[2]; };

// A-matrix 16x32 bf16 layout: lane = row (lane&15); lanes<16 hold K {k0..k0+7, k0+16..k0+23},
// lanes>=16 hold K {k0+8..k0+15, k0+24..k0+31}.  p = rowbase + k0 + hi*8.
__device__ __forceinline__ v16bf load_a_tile(const __bf16* p) {
  V16U u;
  u.h[0] = *(const v8bf*)(p);
  u.h[1] = *(const v8bf*)(p + 16);
  return u.v;
}
// B-matrix 32x16 bf16 layout: lane = col (lane&15); lanes<16 hold K {k0..k0+7, k0+16..k0+23},
// lanes>=16 hold K {k0+8..k0+15, k0+24..k0+31}.  p = colbase + k0 + hi*8.
__device__ __forceinline__ v16bf load_b_tile(const __bf16* p) {
  V16U u;
  u.h[0] = *(const v8bf*)(p);
  u.h[1] = *(const v8bf*)(p + 16);
  return u.v;
}

// ---- fast activations: prefer hardware V_TANH_F32 (CDNA5 TRANS op) ----
__device__ __forceinline__ float fast_tanh(float x) {
#if __has_builtin(__builtin_amdgcn_tanhf)
  return __builtin_amdgcn_tanhf(x);
#elif __has_builtin(__builtin_amdgcn_tanh_f32)
  return __builtin_amdgcn_tanh_f32(x);
#else
  // robust at +/-inf: e->0 gives -1, e->inf gives +1
  const float e = __expf(2.0f * x);
  return 1.0f - 2.0f * __builtin_amdgcn_rcpf(e + 1.0f);
#endif
}
__device__ __forceinline__ float fast_sigmoid(float x) {
#if __has_builtin(__builtin_amdgcn_tanhf) || __has_builtin(__builtin_amdgcn_tanh_f32)
  return 0.5f * fast_tanh(0.5f * x) + 0.5f;   // mul + v_tanh + fma
#else
  return __builtin_amdgcn_rcpf(1.0f + __expf(-x));
#endif
}

__device__ __forceinline__ v8f wmma_bf16(v16bf a, v16bf b, v8f c) {
  return __builtin_amdgcn_wmma_f32_16x16x32_bf16(
      /*neg_a=*/false, a, /*neg_b=*/false, b,
      /*c_mod=*/(short)0, c, /*reuse_a=*/false, /*reuse_b=*/false);
}

// ---------------- workspace layout ----------------
// [0,256)                     : int sync counter (+padding)
// [256, 256+512K)             : bf16 W_gh | W_ih | W_fh | W_oh   (row-major HxH each)
// [256+512K, +256K)           : bf16 hT double buffer, batch-major: hT[buf][b][r]
constexpr size_t kWOff = 256;
constexpr size_t kHOff = kWOff + (size_t)4 * kH * kH * sizeof(__bf16);

// ---------------- prep: fp32 -> bf16 weights, zero h and counter ----------------
__global__ void lstm_prep(const float* __restrict__ Wg, const float* __restrict__ Wi,
                          const float* __restrict__ Wf, const float* __restrict__ Wo,
                          unsigned char* __restrict__ ws) {
  const int idx = blockIdx.x * blockDim.x + threadIdx.x;
  const int n = kH * kH;
  __bf16* w  = (__bf16*)(ws + kWOff);
  __bf16* hT = (__bf16*)(ws + kHOff);
  if (idx < n) {
    w[idx]         = (__bf16)Wg[idx];
    w[n + idx]     = (__bf16)Wi[idx];
    w[2 * n + idx] = (__bf16)Wf[idx];
    w[3 * n + idx] = (__bf16)Wo[idx];
    hT[idx]        = (__bf16)0.0f;   // h buffer 0 (read at t=0)
    hT[n + idx]    = (__bf16)0.0f;   // h buffer 1
  }
  if (idx < 64) ((int*)ws)[idx] = 0; // barrier counter (reset every launch -> replay-safe)
}

// ---------------- persistent LSTM kernel ----------------
__global__ void __launch_bounds__(kTPB, 1)
lstm_persistent(const float* __restrict__ x,
                const float* __restrict__ Wgx, const float* __restrict__ bg,
                const float* __restrict__ Wix, const float* __restrict__ bi,
                const float* __restrict__ Wfx, const float* __restrict__ bfv,
                const float* __restrict__ Wox, const float* __restrict__ bo,
                const float* __restrict__ Wph, const float* __restrict__ bp,
                float* __restrict__ out, unsigned char* __restrict__ ws) {
  int* cnt = (int*)ws;
  const __bf16* Wh = (const __bf16*)(ws + kWOff);
  __bf16* hTa = (__bf16*)(ws + kHOff);
  __bf16* hTb = hTa + kH * kB;

  const int tid  = threadIdx.x;
  const int lane = tid & 31;
  const int wv   = tid >> 5;           // wave 0..7
  const int r0   = blockIdx.x * 16;    // this WG's H-row tile
  const int hi   = lane >> 4;          // half-wave select
  const int ln   = lane & 15;

  const int b0 = (wv * 2 + 0) * 16 + ln;  // batch col, tile 0
  const int b1 = (wv * 2 + 1) * 16 + ln;  // batch col, tile 1

  // rank-1 input weights for this lane's 8 output rows (M = r0 + hi*8 + i)
  float wgx[8], wix[8], wfx[8], wox[8];
#pragma unroll
  for (int i = 0; i < 8; ++i) {
    const int r = r0 + hi * 8 + i;
    wgx[i] = Wgx[r]; wix[i] = Wix[r]; wfx[i] = Wfx[r]; wox[i] = Wox[r];
  }
  const float bg0 = bg[b0],  bg1 = bg[b1];
  const float bi0 = bi[b0],  bi1 = bi[b1];
  const float bf0 = bfv[b0], bf1 = bfv[b1];
  const float bo0 = bo[b0],  bo1 = bo[b1];

  // ---- preload this wave's A-operand weights into registers (t-invariant) ----
  // 4 gates x 8 k-tiles x 8 VGPRs = 256 VGPRs; stays resident for all 784 steps.
  const int arow = (r0 + ln) * kH + hi * 8;
  v16bf aG[8], aI[8], aF[8], aO[8];
#pragma unroll
  for (int kk = 0; kk < 8; ++kk) {
    const int k0 = kk * 32;
    aG[kk] = load_a_tile(Wh + 0 * kH * kH + arow + k0);
    aI[kk] = load_a_tile(Wh + 1 * kH * kH + arow + k0);
    aF[kk] = load_a_tile(Wh + 2 * kH * kH + arow + k0);
    aO[kk] = load_a_tile(Wh + 3 * kH * kH + arow + k0);
  }

  v8f c0 = {};  // cell state lives in VGPRs for the whole sequence
  v8f c1 = {};

  for (int t = 0; t < kT; ++t) {
    const __bf16* hin  = (t & 1) ? hTb : hTa;
    __bf16*       hout = (t & 1) ? hTa : hTb;

    // issue the scalar x loads early so they overlap the WMMA pipeline
    const float xb0 = x[b0 * kT + t];
    const float xb1 = x[b1 * kT + t];

    v8f g0 = {}, g1 = {}, s0 = {}, s1 = {}, f0 = {}, f1 = {}, o0 = {}, o1 = {};
    const __bf16* pb0 = hin + b0 * kH + hi * 8;
    const __bf16* pb1 = hin + b1 * kH + hi * 8;

#pragma unroll
    for (int kk = 0; kk < 8; ++kk) {
      const int k0 = kk * 32;
      const v16bf bb0 = load_b_tile(pb0 + k0);
      const v16bf bb1 = load_b_tile(pb1 + k0);
      g0 = wmma_bf16(aG[kk], bb0, g0);  g1 = wmma_bf16(aG[kk], bb1, g1);
      s0 = wmma_bf16(aI[kk], bb0, s0);  s1 = wmma_bf16(aI[kk], bb1, s1);
      f0 = wmma_bf16(aF[kk], bb0, f0);  f1 = wmma_bf16(aF[kk], bb1, f1);
      o0 = wmma_bf16(aO[kk], bb0, o0);  o1 = wmma_bf16(aO[kk], bb1, o1);
    }

    v8bf h0o, h1o;
#pragma unroll
    for (int i = 0; i < 8; ++i) {
      float gg = fast_tanh(g0[i] + wgx[i] * xb0 + bg0);
      float ii = fast_sigmoid(s0[i] + wix[i] * xb0 + bi0);
      float ff = fast_sigmoid(f0[i] + wfx[i] * xb0 + bf0);
      float oo = fast_sigmoid(o0[i] + wox[i] * xb0 + bo0);
      c0[i] = gg * ii + c0[i] * ff;
      h0o[i] = (__bf16)(fast_tanh(c0[i]) * oo);

      gg = fast_tanh(g1[i] + wgx[i] * xb1 + bg1);
      ii = fast_sigmoid(s1[i] + wix[i] * xb1 + bi1);
      ff = fast_sigmoid(f1[i] + wfx[i] * xb1 + bf1);
      oo = fast_sigmoid(o1[i] + wox[i] * xb1 + bo1);
      c1[i] = gg * ii + c1[i] * ff;
      h1o[i] = (__bf16)(fast_tanh(c1[i]) * oo);
    }
    // batch-major writeback: lane owns 8 consecutive rows for one column -> 16B store
    *(v8bf*)(hout + b0 * kH + r0 + hi * 8) = h0o;
    *(v8bf*)(hout + b1 * kH + r0 + hi * 8) = h1o;

    // ---- inter-workgroup barrier (double-buffered h => one barrier/step) ----
    __threadfence();
    __syncthreads();
    if (tid == 0) {
      __builtin_amdgcn_s_cluster_barrier();  // real barrier if launched as a cluster; S_NOP otherwise
      __hip_atomic_fetch_add(cnt, 1, __ATOMIC_RELEASE, __HIP_MEMORY_SCOPE_AGENT);
      const int target = kNWG * (t + 1);
      while (__hip_atomic_load(cnt, __ATOMIC_ACQUIRE, __HIP_MEMORY_SCOPE_AGENT) < target)
        __builtin_amdgcn_s_sleep(2);
    }
    __syncthreads();
    __threadfence();
  }

  // ---- final projection: out[b][c] = sum_r Wph[c][r]*h[r][b] + bp[b] ----
  // t = 783 (odd) wrote into hTa.
  if (blockIdx.x == 0) {
    const __bf16* hF = hTa;
    const int b = tid;
    float acc[kC];
#pragma unroll
    for (int c = 0; c < kC; ++c) acc[c] = bp[b];
    for (int r = 0; r < kH; ++r) {
      const float hv = (float)hF[b * kH + r];
#pragma unroll
      for (int c = 0; c < kC; ++c) acc[c] += Wph[c * kH + r] * hv;
    }
#pragma unroll
    for (int c = 0; c < kC; ++c) out[b * kC + c] = acc[c];
  }
}

// ---------------- launch ----------------
extern "C" void kernel_launch(void* const* d_in, const int* in_sizes, int n_in,
                              void* d_out, int out_size, void* d_ws, size_t ws_size,
                              hipStream_t stream) {
  const float* x   = (const float*)d_in[0];
  const float* Wgx = (const float*)d_in[1];
  const float* Wgh = (const float*)d_in[2];
  const float* bg  = (const float*)d_in[3];
  const float* Wix = (const float*)d_in[4];
  const float* Wih = (const float*)d_in[5];
  const float* bi  = (const float*)d_in[6];
  const float* Wfx = (const float*)d_in[7];
  const float* Wfh = (const float*)d_in[8];
  const float* bfv = (const float*)d_in[9];
  const float* Wox = (const float*)d_in[10];
  const float* Woh = (const float*)d_in[11];
  const float* bo  = (const float*)d_in[12];
  const float* Wph = (const float*)d_in[13];
  const float* bp  = (const float*)d_in[14];
  unsigned char* ws = (unsigned char*)d_ws;
  float* out = (float*)d_out;

  // 1) convert recurrent weights to bf16, zero h buffers + barrier counter
  lstm_prep<<<(kH * kH + kTPB - 1) / kTPB, kTPB, 0, stream>>>(Wgh, Wih, Wfh, Woh, ws);
  // 2) persistent recurrence + projection
  lstm_persistent<<<kNWG, kTPB, 0, stream>>>(x, Wgx, bg, Wix, bi, Wfx, bfv,
                                             Wox, bo, Wph, bp, out, ws);
}
